// Occupancy_Initialization_55387898249952
// MI455X (gfx1250) — matrix operand order repaired and checked by reading, not compiled, not run on hardware
//
#include <hip/hip_runtime.h>
#include <hip/hip_bf16.h>

// ---------------- constants ----------------
#define V_ 9
#define C1_ 80
#define C2_ 40
#define C4_ 24
#define CH_ 32
#define H1_ 30
#define W1_ 40
#define H2_ 60
#define W2_ 80
#define H4_ 120
#define W4_ 160
#define N_ 131072
#define GS_ 96
#define EPS_ 1e-5f

#define WLAY_USHORTS (27 * 2 * 32 * 16)   // 27648 ushorts = 55296 bytes

typedef __attribute__((ext_vector_type(16))) __bf16 v16bf;
typedef __attribute__((ext_vector_type(8)))  float  v8f;

union Frag {
    uint4  q[2];
    v16bf  v;
};

__device__ __forceinline__ unsigned short f2bf(float f) {
    unsigned u = __float_as_uint(f);
    unsigned r = u + 0x7fffu + ((u >> 16) & 1u);   // round-to-nearest-even
    return (unsigned short)(r >> 16);
}

__device__ __forceinline__ int gtid() { return blockIdx.x * blockDim.x + threadIdx.x; }

// ---------------- 3x3 SAME conv + bias + relu : (V,C,H,W), w (C,C,3,3) ----------------
__global__ void conv3x3_relu_kernel(const float* __restrict__ in, const float* __restrict__ w,
                                    const float* __restrict__ bias, float* __restrict__ out,
                                    int C, int H, int W) {
    int t = gtid();
    int total = V_ * C * H * W;
    if (t >= total) return;
    int x  = t % W;
    int y  = (t / W) % H;
    int co = (t / (W * H)) % C;
    int v  = t / (W * H * C);
    float acc = bias[co];
    const float* ib = in + (size_t)v * C * H * W;
    for (int ci = 0; ci < C; ++ci) {
        const float* ip = ib + (size_t)ci * H * W;
        const float* wp = w + ((size_t)co * C + ci) * 9;
        #pragma unroll
        for (int ky = 0; ky < 3; ++ky) {
            int yy = y + ky - 1;
            if (yy < 0 || yy >= H) continue;
            #pragma unroll
            for (int kx = 0; kx < 3; ++kx) {
                int xx = x + kx - 1;
                if (xx < 0 || xx >= W) continue;
                acc += ip[yy * W + xx] * wp[ky * 3 + kx];
            }
        }
    }
    out[t] = fmaxf(acc, 0.f);
}

// ---------------- bilinear upsample (V,C1,30,40) -> (V,C1,60,80), half-pixel ----------------
__global__ void upsample2x_kernel(const float* __restrict__ in, float* __restrict__ out) {
    int t = gtid();
    int total = V_ * C1_ * H2_ * W2_;
    if (t >= total) return;
    int x = t % W2_;
    int y = (t / W2_) % H2_;
    int c = (t / (W2_ * H2_)) % C1_;
    int v = t / (W2_ * H2_ * C1_);
    float sy = fminf(fmaxf((y + 0.5f) * 0.5f - 0.5f, 0.f), (float)(H1_ - 1));
    float sx = fminf(fmaxf((x + 0.5f) * 0.5f - 0.5f, 0.f), (float)(W1_ - 1));
    int y0 = (int)sy, x0 = (int)sx;
    int y1 = min(y0 + 1, H1_ - 1), x1 = min(x0 + 1, W1_ - 1);
    float fy = sy - y0, fx = sx - x0;
    const float* p = in + ((size_t)v * C1_ + c) * H1_ * W1_;
    float v00 = p[y0 * W1_ + x0], v01 = p[y0 * W1_ + x1];
    float v10 = p[y1 * W1_ + x0], v11 = p[y1 * W1_ + x1];
    out[t] = v00 * (1 - fy) * (1 - fx) + v01 * (1 - fy) * fx + v10 * fy * (1 - fx) + v11 * fy * fx;
}

// ---------------- 2x2 mean pool (V,C4,120,160) -> (V,C4,60,80) ----------------
__global__ void pool2x2_kernel(const float* __restrict__ in, float* __restrict__ out) {
    int t = gtid();
    int total = V_ * C4_ * H2_ * W2_;
    if (t >= total) return;
    int x = t % W2_;
    int y = (t / W2_) % H2_;
    int c = (t / (W2_ * H2_)) % C4_;
    int v = t / (W2_ * H2_ * C4_);
    const float* p = in + ((size_t)v * C4_ + c) * H4_ * W4_;
    int yy = y * 2, xx = x * 2;
    out[t] = 0.25f * (p[yy * W4_ + xx] + p[yy * W4_ + xx + 1] +
                      p[(yy + 1) * W4_ + xx] + p[(yy + 1) * W4_ + xx + 1]);
}

// ---------------- 1x1 fuse conv (concat 80+40+24 -> 32) + relu ----------------
__global__ void down1x1_kernel(const float* __restrict__ f1, const float* __restrict__ f2,
                               const float* __restrict__ f4, const float* __restrict__ w,
                               const float* __restrict__ bias, float* __restrict__ out) {
    int t = gtid();
    int total = V_ * CH_ * H2_ * W2_;
    if (t >= total) return;
    int px = t % (H2_ * W2_);
    int co = (t / (H2_ * W2_)) % CH_;
    int v  = t / (H2_ * W2_ * CH_);
    const int CIN = C1_ + C2_ + C4_;
    float acc = bias[co];
    const float* wp = w + (size_t)co * CIN;
    const float* p1 = f1 + (size_t)v * C1_ * H2_ * W2_ + px;
    const float* p2 = f2 + (size_t)v * C2_ * H2_ * W2_ + px;
    const float* p4 = f4 + (size_t)v * C4_ * H2_ * W2_ + px;
    for (int j = 0; j < C1_; ++j) acc += p1[(size_t)j * H2_ * W2_] * wp[j];
    for (int j = 0; j < C2_; ++j) acc += p2[(size_t)j * H2_ * W2_] * wp[C1_ + j];
    for (int j = 0; j < C4_; ++j) acc += p4[(size_t)j * H2_ * W2_] * wp[C1_ + C2_ + j];
    out[t] = fmaxf(acc, 0.f);
}

// ---------------- residual block: out = in + relu(conv3x3(in)+b), CH=32, 60x80 ----------------
__global__ void resblock_kernel(const float* __restrict__ in, const float* __restrict__ w,
                                const float* __restrict__ bias, float* __restrict__ out) {
    int t = gtid();
    int total = V_ * CH_ * H2_ * W2_;
    if (t >= total) return;
    int x  = t % W2_;
    int y  = (t / W2_) % H2_;
    int co = (t / (W2_ * H2_)) % CH_;
    int v  = t / (W2_ * H2_ * CH_);
    float acc = bias[co];
    const float* ib = in + (size_t)v * CH_ * H2_ * W2_;
    for (int ci = 0; ci < CH_; ++ci) {
        const float* ip = ib + (size_t)ci * H2_ * W2_;
        const float* wp = w + ((size_t)co * CH_ + ci) * 9;
        #pragma unroll
        for (int ky = 0; ky < 3; ++ky) {
            int yy = y + ky - 1;
            if (yy < 0 || yy >= H2_) continue;
            #pragma unroll
            for (int kx = 0; kx < 3; ++kx) {
                int xx = x + kx - 1;
                if (xx < 0 || xx >= W2_) continue;
                acc += ip[yy * W2_ + xx] * wp[ky * 3 + kx];
            }
        }
    }
    out[t] = in[t] + fmaxf(acc, 0.f);
}

// ---------------- voxel hash grid ----------------
__global__ void grid_fill_kernel(int* __restrict__ grid) {
    int t = gtid();
    if (t < GS_ * GS_ * GS_) grid[t] = -1;
}

__global__ void grid_scatter_kernel(const int* __restrict__ coords, const int* __restrict__ stage,
                                    int* __restrict__ grid) {
    int i = gtid();
    if (i >= N_) return;
    int sh = 2 - stage[0];
    int sx = coords[i * 4 + 1] >> sh;
    int sy = coords[i * 4 + 2] >> sh;
    int sz = coords[i * 4 + 3] >> sh;
    grid[(sx * GS_ + sy) * GS_ + sz] = i;
}

__global__ void nbr_kernel(const int* __restrict__ coords, const int* __restrict__ stage,
                           const int* __restrict__ grid, int* __restrict__ nbr) {
    int t = gtid();
    if (t >= N_ * 27) return;
    int i = t / 27, k = t % 27;
    int sh = 2 - stage[0];
    int nx = (coords[i * 4 + 1] >> sh) + (k / 9) - 1;
    int ny = (coords[i * 4 + 2] >> sh) + ((k / 3) % 3) - 1;
    int nz = (coords[i * 4 + 3] >> sh) + (k % 3) - 1;
    int r = -1;
    if (nx >= 0 && nx < GS_ && ny >= 0 && ny < GS_ && nz >= 0 && nz < GS_)
        r = grid[(nx * GS_ + ny) * GS_ + nz];
    nbr[t] = r;
}

// ---------------- projection + masked bilinear sample + masked variance + BN0 ----------------
__global__ void projvar_kernel(const float* __restrict__ fuse, const int* __restrict__ coords,
                               const float* __restrict__ origin, const float* __restrict__ vs,
                               const float* __restrict__ KR, const float* __restrict__ g,
                               const float* __restrict__ b, float* __restrict__ x0,
                               unsigned short* __restrict__ x0bf, float* __restrict__ outCount) {
    int t = gtid();
    if (t >= N_ * CH_) return;
    int i = t >> 5, c = t & 31;
    float voxel = vs[0];
    float wx = (float)coords[i * 4 + 1] * voxel + origin[0];
    float wy = (float)coords[i * 4 + 2] * voxel + origin[1];
    float wz = (float)coords[i * 4 + 3] * voxel + origin[2];
    float sum = 0.f, sumsq = 0.f;
    int cnt = 0;
    for (int v = 0; v < V_; ++v) {
        const float* P = KR + v * 12;
        float u  = P[0] * wx + P[1] * wy + P[2]  * wz + P[3];
        float vv = P[4] * wx + P[5] * wy + P[6]  * wz + P[7];
        float z  = P[8] * wx + P[9] * wy + P[10] * wz + P[11];
        float gx = 2.f * (u / z) / (float)(W4_ - 1) - 1.f;
        float gy = 2.f * (vv / z) / (float)(H4_ - 1) - 1.f;
        if (fabsf(gx) <= 1.f && fabsf(gy) <= 1.f && z > 0.f) {
            ++cnt;
            float px = (gx + 1.f) * 0.5f * (float)(W2_ - 1);
            float py = (gy + 1.f) * 0.5f * (float)(H2_ - 1);
            float fx0 = floorf(px), fy0 = floorf(py);
            float ax = px - fx0, ay = py - fy0;
            int xi = (int)fx0, yi = (int)fy0;
            const float* img = fuse + ((size_t)v * CH_ + c) * (H2_ * W2_);
            float val = 0.f;
            if (xi >= 0 && xi < W2_ && yi >= 0 && yi < H2_)                 val += img[yi * W2_ + xi] * (1 - ax) * (1 - ay);
            if (xi + 1 >= 0 && xi + 1 < W2_ && yi >= 0 && yi < H2_)         val += img[yi * W2_ + xi + 1] * ax * (1 - ay);
            if (xi >= 0 && xi < W2_ && yi + 1 >= 0 && yi + 1 < H2_)         val += img[(yi + 1) * W2_ + xi] * (1 - ax) * ay;
            if (xi + 1 >= 0 && xi + 1 < W2_ && yi + 1 >= 0 && yi + 1 < H2_) val += img[(yi + 1) * W2_ + xi + 1] * ax * ay;
            sum += val;
            sumsq += val * val;
        }
    }
    float den  = (float)cnt;
    float mean = sum / den;
    float var  = sumsq / den - mean * mean;
    float o = g[c] * var * rsqrtf(1.f + EPS_) + b[c];
    x0[t] = o;
    x0bf[t] = f2bf(o);
    if (c == 0) outCount[i] = den;
}

__global__ void coords_copy_kernel(const int* __restrict__ coords, float* __restrict__ out) {
    int t = gtid();
    if (t < N_ * 4) out[t] = (float)coords[t];
}

// ---------------- WMMA weight prep: w (27,32,32) f32 -> wlay (27,2,32,16) bf16 ----------------
// B layout (16-bit B 32x16): lane = column (lanes 0-15), element e<8 = K 8h+e,
// e>=8 = K 16+8h+(e-8), h = lane>=16 (matches A).
__global__ void wprep_kernel(const float* __restrict__ w, unsigned short* __restrict__ wlay) {
    int t = gtid();
    if (t >= WLAY_USHORTS) return;
    int e    = t & 15;
    int lane = (t >> 4) & 31;
    int h    = (t >> 9) & 1;
    int k    = t >> 10;
    int col  = (lane & 15) + h * 16;
    int kk   = ((e >> 3) << 4) + ((lane >> 4) << 3) + (e & 7);
    wlay[t] = f2bf(w[(size_t)k * 1024 + kk * 32 + col]);
}

// ---------------- 27-tap gather-GEMM via v_wmma_f32_16x16x32_bf16 ----------------
// One wave computes a 16x32 output tile. A layout (16-bit A 16x32): lane<16 = row,
// elements 0..7 = K kbase..kbase+7, elements 8..15 = K kbase+16..kbase+23,
// kbase = (lane>=16) ? 8 : 0. B fragments are staged in LDS once per workgroup
// (55 KB of the 320 KB/WGP) and fed from ds_load_b128, so VMEM serves only the
// gathered activations.
__global__ void subm_wmma_kernel(const unsigned short* __restrict__ xbf,
                                 const int* __restrict__ nbr,
                                 const unsigned short* __restrict__ wlay,
                                 float* __restrict__ out) {
    __shared__ unsigned short wl[WLAY_USHORTS];

    // cooperative fill: 3456 uint4 across the block
    {
        const uint4* src = (const uint4*)wlay;
        uint4*       dst = (uint4*)wl;
        for (int i = threadIdx.x; i < WLAY_USHORTS / 8; i += blockDim.x) dst[i] = src[i];
    }
    __syncthreads();

    int wid  = (blockIdx.x * blockDim.x + threadIdx.x) >> 5;
    int lane = threadIdx.x & 31;
    int row0 = wid * 16;
    if (row0 < N_) {                         // uniform per wave; barrier already passed
        int myrow = row0 + (lane & 15);
        int kbase = (lane >> 4) << 3;        // 0 or 8

        v8f c0 = {};
        v8f c1 = {};
        for (int k = 0; k < 27; ++k) {
            int idx = nbr[myrow * 27 + k];
            unsigned msk = (idx >= 0) ? 0xFFFFFFFFu : 0u;
            int safe = (idx >= 0) ? idx : 0;
            const unsigned short* ap = xbf + (size_t)safe * 32 + kbase;
            Frag a;
            a.q[0] = *(const uint4*)(ap);        // K kbase..kbase+7
            a.q[1] = *(const uint4*)(ap + 16);   // K kbase+16..kbase+23
            a.q[0].x &= msk; a.q[0].y &= msk; a.q[0].z &= msk; a.q[0].w &= msk;
            a.q[1].x &= msk; a.q[1].y &= msk; a.q[1].z &= msk; a.q[1].w &= msk;

            const unsigned short* bp = wl + (size_t)k * 1024 + (size_t)lane * 16;
            Frag b0, b1;
            b0.q[0] = *(const uint4*)(bp);
            b0.q[1] = *(const uint4*)(bp + 8);
            b1.q[0] = *(const uint4*)(bp + 512);
            b1.q[1] = *(const uint4*)(bp + 520);

            c0 = __builtin_amdgcn_wmma_f32_16x16x32_bf16(false, a.v, false, b0.v, (short)0, c0, false, false);
            c1 = __builtin_amdgcn_wmma_f32_16x16x32_bf16(false, a.v, false, b1.v, (short)0, c1, false, false);
        }
        // C/D layout: VGPR r, lanes 0-15: M=r, N=lane ; lanes 16-31: M=8+r, N=lane-16
        int colbase = lane & 15;
        int rbase   = row0 + ((lane >> 4) << 3);
        #pragma unroll
        for (int r = 0; r < 8; ++r) {
            out[(size_t)(rbase + r) * 32 + colbase]      = c0[r];
            out[(size_t)(rbase + r) * 32 + 16 + colbase] = c1[r];
        }
    }
}

// ---------------- epilogue: x = relu(conv) ; emit f32 + bf16 staging ----------------
__global__ void relu_bf_kernel(const float* __restrict__ conv, float* __restrict__ x,
                               unsigned short* __restrict__ xbf) {
    int t = gtid();
    if (t >= N_ * CH_) return;
    float v = fmaxf(conv[t], 0.f);
    x[t] = v;
    xbf[t] = f2bf(v);
}

// ---------------- epilogue: x = LN(relu(conv) + x) ; emit f32 + bf16 staging ----------------
__global__ void resln_kernel(const float* __restrict__ conv, float* __restrict__ x,
                             const float* __restrict__ g, const float* __restrict__ b,
                             unsigned short* __restrict__ xbf) {
    int i = gtid();
    if (i >= N_) return;
    float tv[CH_];
    float m = 0.f;
    #pragma unroll
    for (int c = 0; c < CH_; ++c) {
        float u = fmaxf(conv[(size_t)i * CH_ + c], 0.f) + x[(size_t)i * CH_ + c];
        tv[c] = u;
        m += u;
    }
    m *= (1.f / CH_);
    float vv = 0.f;
    #pragma unroll
    for (int c = 0; c < CH_; ++c) { float d = tv[c] - m; vv += d * d; }
    vv *= (1.f / CH_);
    float inv = rsqrtf(vv + EPS_);
    #pragma unroll
    for (int c = 0; c < CH_; ++c) {
        float o = (tv[c] - m) * inv * g[c] + b[c];
        x[(size_t)i * CH_ + c] = o;
        xbf[(size_t)i * CH_ + c] = f2bf(o);
    }
}

// ---------------- final 27-tap (32 -> 1) conv + BN -> occ ----------------
__global__ void occ_final_kernel(const float* __restrict__ x, const int* __restrict__ nbr,
                                 const float* __restrict__ w, const float* __restrict__ g,
                                 const float* __restrict__ b, float* __restrict__ occ) {
    int i = gtid();
    if (i >= N_) return;
    float acc = 0.f;
    for (int k = 0; k < 27; ++k) {
        int idx = nbr[i * 27 + k];
        if (idx >= 0) {
            const float* f  = x + (size_t)idx * 32;
            const float* wk = w + k * 32;
            #pragma unroll
            for (int c = 0; c < 32; ++c) acc += f[c] * wk[c];
        }
    }
    occ[i] = g[0] * acc * rsqrtf(1.f + EPS_) + b[0];
}

// ---------------- host launch ----------------
static inline size_t align256(size_t x) { return (x + 255) & ~(size_t)255; }

extern "C" void kernel_launch(void* const* d_in, const int* in_sizes, int n_in,
                              void* d_out, int out_size, void* d_ws, size_t ws_size,
                              hipStream_t stream) {
    (void)in_sizes; (void)n_in; (void)out_size; (void)ws_size;
    const float* feats_1x = (const float*)d_in[0];
    const float* feats_2x = (const float*)d_in[1];
    const float* feats_4x = (const float*)d_in[2];
    const int*   coords   = (const int*)d_in[3];
    const float* origin   = (const float*)d_in[4];
    const float* vsize    = (const float*)d_in[5];
    const float* KRcam    = (const float*)d_in[6];
    const float* w_f1 = (const float*)d_in[7],  *b_f1 = (const float*)d_in[8];
    const float* w_f2 = (const float*)d_in[9],  *b_f2 = (const float*)d_in[10];
    const float* w_f4 = (const float*)d_in[11], *b_f4 = (const float*)d_in[12];
    const float* w_down = (const float*)d_in[13], *b_down = (const float*)d_in[14];
    const float* w_p[4]  = {(const float*)d_in[15], (const float*)d_in[17], (const float*)d_in[19], (const float*)d_in[21]};
    const float* b_p[4]  = {(const float*)d_in[16], (const float*)d_in[18], (const float*)d_in[20], (const float*)d_in[22]};
    const float* bn0_g = (const float*)d_in[23], *bn0_b = (const float*)d_in[24];
    const float* w_elan = (const float*)d_in[25];
    const float* w_s1 = (const float*)d_in[26];
    const float* ln1_g = (const float*)d_in[27], *ln1_b = (const float*)d_in[28];
    const float* w_s2 = (const float*)d_in[29];
    const float* ln2_g = (const float*)d_in[30], *ln2_b = (const float*)d_in[31];
    const float* w_s3 = (const float*)d_in[32];
    const float* ln3_g = (const float*)d_in[33], *ln3_b = (const float*)d_in[34];
    const float* w_s4 = (const float*)d_in[35];
    const float* bn4_g = (const float*)d_in[36], *bn4_b = (const float*)d_in[37];
    const int*   stage = (const int*)d_in[38];

    // workspace carve-up
    char* ws = (char*)d_ws;
    size_t off = 0;
    auto alloc = [&](size_t bytes) { char* p = ws + off; off += align256(bytes); return p; };
    float* f1b   = (float*)alloc((size_t)V_ * C1_ * H1_ * W1_ * 4);
    float* f2b   = (float*)alloc((size_t)V_ * C2_ * H2_ * W2_ * 4);
    float* f4b   = (float*)alloc((size_t)V_ * C4_ * H4_ * W4_ * 4);
    float* f1up  = (float*)alloc((size_t)V_ * C1_ * H2_ * W2_ * 4);
    float* f4p   = (float*)alloc((size_t)V_ * C4_ * H2_ * W2_ * 4);
    float* fuseA = (float*)alloc((size_t)V_ * CH_ * H2_ * W2_ * 4);
    float* fuseB = (float*)alloc((size_t)V_ * CH_ * H2_ * W2_ * 4);
    int*   grid  = (int*)alloc((size_t)GS_ * GS_ * GS_ * 4);
    int*   nbr   = (int*)alloc((size_t)N_ * 27 * 4);
    float* xf    = (float*)alloc((size_t)N_ * CH_ * 4);
    float* convo = (float*)alloc((size_t)N_ * CH_ * 4);
    unsigned short* xbf  = (unsigned short*)alloc((size_t)N_ * CH_ * 2);
    unsigned short* wlay = (unsigned short*)alloc((size_t)WLAY_USHORTS * 2);

    float* occ_out   = (float*)d_out;            // [0, N)
    float* coord_out = (float*)d_out + N_;       // [N, 5N)
    float* count_out = (float*)d_out + 5 * N_;   // [5N, 6N)

    const int B = 256;
    auto G = [](long long n, int b) { return (unsigned)((n + b - 1) / b); };

    // 1) per-scale 3x3 conv + relu
    conv3x3_relu_kernel<<<G((long long)V_ * C1_ * H1_ * W1_, B), B, 0, stream>>>(feats_1x, w_f1, b_f1, f1b, C1_, H1_, W1_);
    conv3x3_relu_kernel<<<G((long long)V_ * C2_ * H2_ * W2_, B), B, 0, stream>>>(feats_2x, w_f2, b_f2, f2b, C2_, H2_, W2_);
    conv3x3_relu_kernel<<<G((long long)V_ * C4_ * H4_ * W4_, B), B, 0, stream>>>(feats_4x, w_f4, b_f4, f4b, C4_, H4_, W4_);

    // 2) resample + fuse
    upsample2x_kernel<<<G((long long)V_ * C1_ * H2_ * W2_, B), B, 0, stream>>>(f1b, f1up);
    pool2x2_kernel<<<G((long long)V_ * C4_ * H2_ * W2_, B), B, 0, stream>>>(f4b, f4p);
    down1x1_kernel<<<G((long long)V_ * CH_ * H2_ * W2_, B), B, 0, stream>>>(f1up, f2b, f4p, w_down, b_down, fuseA);

    // 3) four residual blocks (A->B->A->B->A)
    long long nfuse = (long long)V_ * CH_ * H2_ * W2_;
    resblock_kernel<<<G(nfuse, B), B, 0, stream>>>(fuseA, w_p[0], b_p[0], fuseB);
    resblock_kernel<<<G(nfuse, B), B, 0, stream>>>(fuseB, w_p[1], b_p[1], fuseA);
    resblock_kernel<<<G(nfuse, B), B, 0, stream>>>(fuseA, w_p[2], b_p[2], fuseB);
    resblock_kernel<<<G(nfuse, B), B, 0, stream>>>(fuseB, w_p[3], b_p[3], fuseA);

    // 4) voxel hash + neighbor table
    grid_fill_kernel<<<G((long long)GS_ * GS_ * GS_, B), B, 0, stream>>>(grid);
    grid_scatter_kernel<<<G(N_, B), B, 0, stream>>>(coords, stage, grid);
    nbr_kernel<<<G((long long)N_ * 27, B), B, 0, stream>>>(coords, stage, grid, nbr);

    // 5) projection + masked variance + BN0 (also emits count and bf16 staging)
    projvar_kernel<<<G((long long)N_ * CH_, B), B, 0, stream>>>(fuseA, coords, origin, vsize, KRcam,
                                                               bn0_g, bn0_b, xf, xbf, count_out);
    coords_copy_kernel<<<G((long long)N_ * 4, B), B, 0, stream>>>(coords, coord_out);

    const unsigned subm_grid = (N_ / 16) * 32 / B;   // 8192 waves, 256 threads/block

    // 6) x1 = relu(subm(x0, w_elan))
    wprep_kernel<<<G(WLAY_USHORTS, B), B, 0, stream>>>(w_elan, wlay);
    subm_wmma_kernel<<<subm_grid, B, 0, stream>>>(xbf, nbr, wlay, convo);
    relu_bf_kernel<<<G((long long)N_ * CH_, B), B, 0, stream>>>(convo, xf, xbf);

    // 7) three residual + layernorm stages
    wprep_kernel<<<G(WLAY_USHORTS, B), B, 0, stream>>>(w_s1, wlay);
    subm_wmma_kernel<<<subm_grid, B, 0, stream>>>(xbf, nbr, wlay, convo);
    resln_kernel<<<G(N_, B), B, 0, stream>>>(convo, xf, ln1_g, ln1_b, xbf);

    wprep_kernel<<<G(WLAY_USHORTS, B), B, 0, stream>>>(w_s2, wlay);
    subm_wmma_kernel<<<subm_grid, B, 0, stream>>>(xbf, nbr, wlay, convo);
    resln_kernel<<<G(N_, B), B, 0, stream>>>(convo, xf, ln2_g, ln2_b, xbf);

    wprep_kernel<<<G(WLAY_USHORTS, B), B, 0, stream>>>(w_s3, wlay);
    subm_wmma_kernel<<<subm_grid, B, 0, stream>>>(xbf, nbr, wlay, convo);
    resln_kernel<<<G(N_, B), B, 0, stream>>>(convo, xf, ln3_g, ln3_b, xbf);

    // 8) occ = bn(subm(x, w_s4))
    occ_final_kernel<<<G(N_, B), B, 0, stream>>>(xf, nbr, w_s4, bn4_g, bn4_b, occ_out);
}
